// EncProcDecGNN_84688165143203
// MI455X (gfx1250) — matrix operand order repaired and checked by reading, not compiled, not run on hardware
//
#include <hip/hip_runtime.h>
#include <hip/hip_bf16.h>

typedef __attribute__((ext_vector_type(16))) _Float16 v16h;
typedef __attribute__((ext_vector_type(8)))  float    v8f;
typedef __attribute__((ext_vector_type(2)))  _Float16 v2h;
typedef __attribute__((ext_vector_type(4)))  int      v4i;

#define AS1 __attribute__((address_space(1)))
#define AS3 __attribute__((address_space(3)))

#define N_NODES 40000
#define N_EDGES 640000
#define ND_IN   64
#define ED_IN   32
#define OUT_D   3
#define HDIM    128
#define MSTEPS  10
#define LN_EPS  1e-5f

#if __has_builtin(__builtin_amdgcn_global_load_async_to_lds_b128)
#define HAVE_ASYNC_LDS 1
#else
#define HAVE_ASYNC_LDS 0
#endif

__device__ __forceinline__ void wait_async0() {
#if HAVE_ASYNC_LDS
#if __has_builtin(__builtin_amdgcn_s_wait_asynccnt)
  __builtin_amdgcn_s_wait_asynccnt(0);
#else
  asm volatile("s_wait_asynccnt 0x0" ::: "memory");
#endif
#endif
}

// Packed f16 atomic add (segment_sum scatter). Falls back to CAS if the
// fadd builtin is unavailable.
__device__ __forceinline__ void atomic_pk_add_f16(_Float16* p, float y0, float y1) {
#if __has_builtin(__builtin_amdgcn_global_atomic_fadd_v2f16)
  v2h v; v[0] = (_Float16)y0; v[1] = (_Float16)y1;
  __builtin_amdgcn_global_atomic_fadd_v2f16((AS1 v2h*)p, v);
#else
  unsigned int* u = (unsigned int*)p;
  unsigned int old = *u, assumed;
  do {
    assumed = old;
    union { unsigned int w; _Float16 h[2]; } cur, nxt;
    cur.w = assumed;
    nxt.h[0] = (_Float16)((float)cur.h[0] + y0);
    nxt.h[1] = (_Float16)((float)cur.h[1] + y1);
    old = atomicCAS(u, assumed, nxt.w);
  } while (old != assumed);
#endif
}

// ---------------------------------------------------------------------------
// Weight convert: W [steps, K, N] f32 row-major -> Wt [steps, 128, K] f16
// (column-major per step, padded to 128 output columns with zeros)
// ---------------------------------------------------------------------------
__global__ void convert_w_kernel(const float* __restrict__ W,
                                 _Float16* __restrict__ Wt,
                                 int K, int N, long total) {
  long i = (long)blockIdx.x * blockDim.x + threadIdx.x;
  if (i >= total) return;
  int  k   = (int)(i % K);
  long rem = i / K;
  int  n   = (int)(rem % 128);
  long s   = rem / 128;
  float v = (n < N) ? W[(s * K + k) * N + n] : 0.0f;
  Wt[i] = (_Float16)v;
}

__global__ void zero_kernel(float* __restrict__ p, long n) {
  long i = (long)blockIdx.x * blockDim.x + threadIdx.x;
  if (i < n) p[i] = 0.0f;
}

// ---------------------------------------------------------------------------
// Fused 3-layer MLP (+optional LayerNorm, residual, scatter-add) over a
// 64-row tile. 256 threads = 8 waves; each wave computes a 16x64 output tile
// with 4x v_wmma_f32_16x16x32_f16 per K-chunk of 32.
// ---------------------------------------------------------------------------
struct MlpArgs {
  const void* x0; const void* x1; const void* x2;      // input segments
  const int* idx0; const int* idx1; const int* idx2;   // row gather (null=id)
  int c0, c1, c2;                                      // segment widths
  int xHalf;                                           // 1: segments are f16
  const _Float16* Wt0; const _Float16* Wt1; const _Float16* Wt2; // [128][K]
  const float* b0; const float* b1; const float* b2;
  const float* g;  const float* bn;                    // LN params (null=none)
  void* out; int outCols; int residual;                // f16 out (f32 if !g)
  _Float16* agg; const int* aggIdx;                    // pk-f16 scatter-add
  int nRows;                                           // multiple of 64
};

__device__ __forceinline__ void tile_gemm(const _Float16* __restrict__ A, int K,
                                          const _Float16* __restrict__ Wt,
                                          const float* __restrict__ bias,
                                          int biasCols,
                                          _Float16* dstH, float* dstF, int tid) {
  const int lane = tid & 31;
  const int wid  = tid >> 5;
  const int wr   = wid & 3;     // 4 row-tiles of 16 rows
  const int wc   = wid >> 2;    // 2 col-halves of 64 cols
  const int half = lane >> 4;
  const int l    = lane & 15;

  __builtin_prefetch(Wt + (long)(wc * 64 + l) * K, 0, 0);  // global_prefetch_b8

  v8f acc[4] = {};
  for (int k0 = 0; k0 < K; k0 += 32) {
    // A fragment 16x32 f16: lane<16 holds row l, K {k0..+15};
    // lane>=16 holds K {k0+16..+31}.  Two aligned 16B LDS reads.
    v16h a;
    const _Float16* ap = A + (wr * 16 + l) * K + k0 + half * 16;
#pragma unroll
    for (int i = 0; i < 8; ++i) { a[i] = ap[i]; a[i + 8] = ap[8 + i]; }
#pragma unroll
    for (int t = 0; t < 4; ++t) {
      // B fragment 32x16: col-major weights -> one contiguous 32B load.
      const int col = wc * 64 + t * 16 + l;
      v16h b = *(const v16h*)(Wt + (long)col * K + k0 + half * 16);
      acc[t] = __builtin_amdgcn_wmma_f32_16x16x32_f16(
          false, a, false, b, (short)0, acc[t], false, false);
    }
  }
  // C layout: VGPR j -> row (wr*16 + half*8 + j), col (wc*64 + t*16 + l)
#pragma unroll
  for (int t = 0; t < 4; ++t) {
    const int col = wc * 64 + t * 16 + l;
    const float bv = (col < biasCols) ? bias[col] : 0.0f;
#pragma unroll
    for (int j = 0; j < 8; ++j) {
      const int row = wr * 16 + half * 8 + j;
      float v = acc[t][j] + bv;
      if (dstF) {
        dstF[row * HDIM + col] = v;               // final layer: raw f32
      } else {
        v = v > 0.0f ? v : 0.0f;                  // hidden layer: ReLU -> f16
        dstH[row * HDIM + col] = (_Float16)v;
      }
    }
  }
}

__global__ __launch_bounds__(256)
void fused_mlp_kernel(MlpArgs a) {
  __shared__ __align__(16) char smem[65536];
  _Float16* ldsX   = (_Float16*)smem;             // 64 x K (K<=384)  : 48KB max
  _Float16* ldsH1  = (_Float16*)(smem + 49152);   // 64 x 128 f16     : 16KB
  _Float16* ldsH2  = (_Float16*)smem;             // reuse input slot : 16KB
  float*    ldsOut = (float*)(smem + 16384);      // 64 x 128 f32     : 32KB

  const int tid = threadIdx.x;
  const int rowBase = blockIdx.x * 64;
  const int K = a.c0 + a.c1 + a.c2;

  if (a.xHalf) {
    // f16 segments: stage the gather as 16B async global->LDS copies.
    const int chunks = K >> 3;                    // 8 halves per copy
    for (int i = tid; i < 64 * chunks; i += 256) {
      const int row = i / chunks;
      const int c   = (i - row * chunks) << 3;
      const int r   = rowBase + row;
      const _Float16* xp; const int* idxp; int coff, cw;
      if (c < a.c0)             { xp = (const _Float16*)a.x0; idxp = a.idx0; coff = c;               cw = a.c0; }
      else if (c < a.c0 + a.c1) { xp = (const _Float16*)a.x1; idxp = a.idx1; coff = c - a.c0;        cw = a.c1; }
      else                      { xp = (const _Float16*)a.x2; idxp = a.idx2; coff = c - a.c0 - a.c1; cw = a.c2; }
      const int rr = idxp ? idxp[r] : r;
      const _Float16* gsrc = xp + (long)rr * cw + coff;
      _Float16* ldst = ldsX + row * K + c;
#if HAVE_ASYNC_LDS
      __builtin_amdgcn_global_load_async_to_lds_b128(
          (AS1 v4i*)gsrc, (AS3 v4i*)ldst, 0, 0);
#else
      *(v4i*)ldst = *(const v4i*)gsrc;
#endif
    }
    wait_async0();
  } else {
    // f32 segments (encoder inputs): vectorized load + convert to f16.
    const int chunks = K >> 2;                    // 4 floats per step
    for (int i = tid; i < 64 * chunks; i += 256) {
      const int row = i / chunks;
      const int c   = (i - row * chunks) << 2;
      const int r   = rowBase + row;
      const float* xp; const int* idxp; int coff, cw;
      if (c < a.c0)             { xp = (const float*)a.x0; idxp = a.idx0; coff = c;               cw = a.c0; }
      else if (c < a.c0 + a.c1) { xp = (const float*)a.x1; idxp = a.idx1; coff = c - a.c0;        cw = a.c1; }
      else                      { xp = (const float*)a.x2; idxp = a.idx2; coff = c - a.c0 - a.c1; cw = a.c2; }
      const int rr = idxp ? idxp[r] : r;
      const float4 v = *(const float4*)(xp + (long)rr * cw + coff);
      _Float16* d = ldsX + row * K + c;
      d[0] = (_Float16)v.x; d[1] = (_Float16)v.y;
      d[2] = (_Float16)v.z; d[3] = (_Float16)v.w;
    }
  }
  __syncthreads();

  tile_gemm(ldsX,  K,    a.Wt0, a.b0, HDIM,      ldsH1, nullptr, tid);
  __syncthreads();
  tile_gemm(ldsH1, HDIM, a.Wt1, a.b1, HDIM,      ldsH2, nullptr, tid);
  __syncthreads();
  tile_gemm(ldsH2, HDIM, a.Wt2, a.b2, a.outCols, nullptr, ldsOut, tid);
  __syncthreads();

  // Epilogue: 4 threads per row (consecutive lanes); mean/var reduced with
  // __shfl_xor within the 4-lane group.
  const int row  = tid >> 2;
  const int part = tid & 3;
  const int r    = rowBase + row;
  const float* rowv = ldsOut + row * HDIM;
  const int cBeg = part * 32, cEnd = cBeg + 32;

  if (a.g == nullptr) {
    // Decoder: raw f32 output, outCols (=3) columns.
    if (part == 0) {
      float* o = (float*)a.out;
      for (int c = 0; c < a.outCols; ++c)
        o[(long)r * a.outCols + c] = rowv[c];
    }
    return;
  }

  float s = 0.0f;
  for (int c = cBeg; c < cEnd; ++c) s += rowv[c];
  s += __shfl_xor(s, 1); s += __shfl_xor(s, 2);
  const float mu = s * (1.0f / HDIM);
  float q = 0.0f;
  for (int c = cBeg; c < cEnd; ++c) { float d = rowv[c] - mu; q += d * d; }
  q += __shfl_xor(q, 1); q += __shfl_xor(q, 2);
  const float rs = rsqrtf(q * (1.0f / HDIM) + LN_EPS);

  _Float16* o = (_Float16*)a.out;
  const long base = (long)r * HDIM;
  if (a.agg) {
    const long abase = (long)a.aggIdx[r] * HDIM;
    for (int c = cBeg; c < cEnd; c += 2) {
      const float y0 = (rowv[c]     - mu) * rs * a.g[c]     + a.bn[c];
      const float y1 = (rowv[c + 1] - mu) * rs * a.g[c + 1] + a.bn[c + 1];
      o[base + c]     = (_Float16)((float)o[base + c]     + y0);   // e += e_new
      o[base + c + 1] = (_Float16)((float)o[base + c + 1] + y1);
      atomic_pk_add_f16(a.agg + abase + c, y0, y1);                // segment_sum
    }
  } else if (a.residual) {
    for (int c = cBeg; c < cEnd; ++c) {
      const float y = (rowv[c] - mu) * rs * a.g[c] + a.bn[c];
      o[base + c] = (_Float16)((float)o[base + c] + y);            // n += n_new
    }
  } else {
    for (int c = cBeg; c < cEnd; ++c) {
      const float y = (rowv[c] - mu) * rs * a.g[c] + a.bn[c];
      o[base + c] = (_Float16)y;                                   // encoder
    }
  }
}

// ---------------------------------------------------------------------------
// Host orchestration
// ---------------------------------------------------------------------------
extern "C" void kernel_launch(void* const* d_in, const int* in_sizes, int n_in,
                              void* d_out, int out_size, void* d_ws, size_t ws_size,
                              hipStream_t stream) {
  // jax flattens dicts in sorted key order: W0,W1,W2,b0,b1,b2[,bn,g]
  const float* nodes     = (const float*)d_in[0];
  const float* edge_attr = (const float*)d_in[1];
  const float* en_W[3] = {(const float*)d_in[2],  (const float*)d_in[3],  (const float*)d_in[4]};
  const float* en_b[3] = {(const float*)d_in[5],  (const float*)d_in[6],  (const float*)d_in[7]};
  const float* en_bn   =  (const float*)d_in[8];
  const float* en_g    =  (const float*)d_in[9];
  const float* ee_W[3] = {(const float*)d_in[10], (const float*)d_in[11], (const float*)d_in[12]};
  const float* ee_b[3] = {(const float*)d_in[13], (const float*)d_in[14], (const float*)d_in[15]};
  const float* ee_bn   =  (const float*)d_in[16];
  const float* ee_g    =  (const float*)d_in[17];
  const float* pe_W[3] = {(const float*)d_in[18], (const float*)d_in[19], (const float*)d_in[20]};
  const float* pe_b[3] = {(const float*)d_in[21], (const float*)d_in[22], (const float*)d_in[23]};
  const float* pe_bn   =  (const float*)d_in[24];
  const float* pe_g    =  (const float*)d_in[25];
  const float* pn_W[3] = {(const float*)d_in[26], (const float*)d_in[27], (const float*)d_in[28]};
  const float* pn_b[3] = {(const float*)d_in[29], (const float*)d_in[30], (const float*)d_in[31]};
  const float* pn_bn   =  (const float*)d_in[32];
  const float* pn_g    =  (const float*)d_in[33];
  const float* de_W[3] = {(const float*)d_in[34], (const float*)d_in[35], (const float*)d_in[36]};
  const float* de_b[3] = {(const float*)d_in[37], (const float*)d_in[38], (const float*)d_in[39]};
  const int* edge_index = (const int*)d_in[40];
  const int* srcIdx = edge_index;
  const int* dstIdx = edge_index + N_EDGES;

  // Workspace carve-up (f16 graph state: halves HBM traffic vs f32)
  char* ws = (char*)d_ws;
  size_t off = 0;
  auto carve = [&](size_t bytes) -> void* {
    void* p = ws + off;
    off = (off + bytes + 255) & ~(size_t)255;
    return p;
  };
  _Float16* nBuf = (_Float16*)carve((size_t)N_NODES * HDIM * 2);
  _Float16* eBuf = (_Float16*)carve((size_t)N_EDGES * HDIM * 2);
  _Float16* agg  = (_Float16*)carve((size_t)N_NODES * HDIM * 2);
  auto carveW = [&](int K, int steps) -> _Float16* {
    return (_Float16*)carve((size_t)steps * 128 * K * 2);
  };
  _Float16* wt_en[3] = {carveW(ND_IN, 1), carveW(HDIM, 1), carveW(HDIM, 1)};
  _Float16* wt_ee[3] = {carveW(ED_IN, 1), carveW(HDIM, 1), carveW(HDIM, 1)};
  _Float16* wt_pe[3] = {carveW(3 * HDIM, MSTEPS), carveW(HDIM, MSTEPS), carveW(HDIM, MSTEPS)};
  _Float16* wt_pn[3] = {carveW(2 * HDIM, MSTEPS), carveW(HDIM, MSTEPS), carveW(HDIM, MSTEPS)};
  _Float16* wt_de[3] = {carveW(HDIM, 1), carveW(HDIM, 1), carveW(HDIM, 1)};
  (void)ws_size; (void)in_sizes; (void)n_in; (void)out_size;

  auto cvt = [&](const float* W, _Float16* Wt, int K, int N, int steps) {
    long total = (long)steps * 128 * K;
    int blocks = (int)((total + 255) / 256);
    convert_w_kernel<<<blocks, 256, 0, stream>>>(W, Wt, K, N, total);
  };
  cvt(en_W[0], wt_en[0], ND_IN, HDIM, 1);
  cvt(en_W[1], wt_en[1], HDIM,  HDIM, 1);
  cvt(en_W[2], wt_en[2], HDIM,  HDIM, 1);
  cvt(ee_W[0], wt_ee[0], ED_IN, HDIM, 1);
  cvt(ee_W[1], wt_ee[1], HDIM,  HDIM, 1);
  cvt(ee_W[2], wt_ee[2], HDIM,  HDIM, 1);
  cvt(pe_W[0], wt_pe[0], 3 * HDIM, HDIM, MSTEPS);
  cvt(pe_W[1], wt_pe[1], HDIM,     HDIM, MSTEPS);
  cvt(pe_W[2], wt_pe[2], HDIM,     HDIM, MSTEPS);
  cvt(pn_W[0], wt_pn[0], 2 * HDIM, HDIM, MSTEPS);
  cvt(pn_W[1], wt_pn[1], HDIM,     HDIM, MSTEPS);
  cvt(pn_W[2], wt_pn[2], HDIM,     HDIM, MSTEPS);
  cvt(de_W[0], wt_de[0], HDIM, HDIM, 1);
  cvt(de_W[1], wt_de[1], HDIM, HDIM, 1);
  cvt(de_W[2], wt_de[2], HDIM, OUT_D, 1);

  const int nodeBlocks = N_NODES / 64;   // exact
  const int edgeBlocks = N_EDGES / 64;   // exact

  MlpArgs a;
  // ---- Encoder: node MLP ----
  a = {};
  a.x0 = nodes; a.c0 = ND_IN; a.xHalf = 0;
  a.Wt0 = wt_en[0]; a.Wt1 = wt_en[1]; a.Wt2 = wt_en[2];
  a.b0 = en_b[0]; a.b1 = en_b[1]; a.b2 = en_b[2];
  a.g = en_g; a.bn = en_bn;
  a.out = nBuf; a.outCols = HDIM; a.nRows = N_NODES;
  fused_mlp_kernel<<<nodeBlocks, 256, 0, stream>>>(a);

  // ---- Encoder: edge MLP ----
  a = {};
  a.x0 = edge_attr; a.c0 = ED_IN; a.xHalf = 0;
  a.Wt0 = wt_ee[0]; a.Wt1 = wt_ee[1]; a.Wt2 = wt_ee[2];
  a.b0 = ee_b[0]; a.b1 = ee_b[1]; a.b2 = ee_b[2];
  a.g = ee_g; a.bn = ee_bn;
  a.out = eBuf; a.outCols = HDIM; a.nRows = N_EDGES;
  fused_mlp_kernel<<<edgeBlocks, 256, 0, stream>>>(a);

  // ---- Processor: MSTEPS interaction networks ----
  for (int s = 0; s < MSTEPS; ++s) {
    long aggN = (long)N_NODES * HDIM / 2;   // f16 zero via f32 stores
    zero_kernel<<<(int)((aggN + 255) / 256), 256, 0, stream>>>((float*)agg, aggN);

    // edge update: e_new = MLP([e | n[src] | n[dst]]); e += e_new; agg[dst] += e_new
    a = {};
    a.x0 = eBuf; a.c0 = HDIM;
    a.x1 = nBuf; a.idx1 = srcIdx; a.c1 = HDIM;
    a.x2 = nBuf; a.idx2 = dstIdx; a.c2 = HDIM;
    a.xHalf = 1;
    a.Wt0 = wt_pe[0] + (size_t)s * 128 * (3 * HDIM);
    a.Wt1 = wt_pe[1] + (size_t)s * 128 * HDIM;
    a.Wt2 = wt_pe[2] + (size_t)s * 128 * HDIM;
    a.b0 = pe_b[0] + s * HDIM; a.b1 = pe_b[1] + s * HDIM; a.b2 = pe_b[2] + s * HDIM;
    a.g = pe_g + s * HDIM; a.bn = pe_bn + s * HDIM;
    a.out = eBuf; a.outCols = HDIM; a.residual = 1;
    a.agg = agg; a.aggIdx = dstIdx;
    a.nRows = N_EDGES;
    fused_mlp_kernel<<<edgeBlocks, 256, 0, stream>>>(a);

    // node update: n_new = MLP([n | agg]); n += n_new
    a = {};
    a.x0 = nBuf; a.c0 = HDIM;
    a.x1 = agg;  a.c1 = HDIM;
    a.xHalf = 1;
    a.Wt0 = wt_pn[0] + (size_t)s * 128 * (2 * HDIM);
    a.Wt1 = wt_pn[1] + (size_t)s * 128 * HDIM;
    a.Wt2 = wt_pn[2] + (size_t)s * 128 * HDIM;
    a.b0 = pn_b[0] + s * HDIM; a.b1 = pn_b[1] + s * HDIM; a.b2 = pn_b[2] + s * HDIM;
    a.g = pn_g + s * HDIM; a.bn = pn_bn + s * HDIM;
    a.out = nBuf; a.outCols = HDIM; a.residual = 1;
    a.nRows = N_NODES;
    fused_mlp_kernel<<<nodeBlocks, 256, 0, stream>>>(a);
  }

  // ---- Decoder (no LayerNorm, f32 output) ----
  a = {};
  a.x0 = nBuf; a.c0 = HDIM; a.xHalf = 1;
  a.Wt0 = wt_de[0]; a.Wt1 = wt_de[1]; a.Wt2 = wt_de[2];
  a.b0 = de_b[0]; a.b1 = de_b[1]; a.b2 = de_b[2];
  a.g = nullptr; a.bn = nullptr;
  a.out = d_out; a.outCols = OUT_D; a.nRows = N_NODES;
  fused_mlp_kernel<<<nodeBlocks, 256, 0, stream>>>(a);
}
